// DiscRNNGrammar_541165879739
// MI455X (gfx1250) — compile-verified, matrix operand order checked by reading, not run on hardware
//
#include <hip/hip_runtime.h>

// ---------------------------------------------------------------------------
// DiscRNNGrammar forward for MI455X (gfx1250), compile-only tuning build.
//
// fp8 WMMA recurrent LSTM with LDS-resident weights (256KB) + h state (16KB)
// + double-buffered async x staging (16KB) = 288KB of the 320KB WGP LDS.
// Cell state lives in registers across all ~1025 timesteps. x_{t+1} is
// brought in with GLOBAL_LOAD_ASYNC_TO_LDS_B128 (ASYNCcnt) overlapped with
// step t's WMMAs, when the toolchain exposes the async builtins.
// ---------------------------------------------------------------------------

#define BB    64
#define LL    1024
#define TT    1026      // L + 2
#define HID   128
#define GATES 512
#define NTOP  1027      // stack_tops rows == feats rows per sentence
#define MROWS (BB * NTOP)  // 65728, multiple of 64

typedef __attribute__((ext_vector_type(8))) int   v8i;
typedef __attribute__((ext_vector_type(8))) float v8f;
typedef __attribute__((ext_vector_type(4))) int   i32x4;

#if __has_builtin(__builtin_amdgcn_global_load_async_to_lds_b128) && \
    __has_builtin(__builtin_amdgcn_s_wait_asynccnt)
#define USE_ASYNC_X 1
#else
#define USE_ASYNC_X 0
#endif

#if USE_ASYNC_X
typedef __attribute__((address_space(1))) i32x4 g_i32x4;
typedef __attribute__((address_space(3))) i32x4 l_i32x4;
__device__ __forceinline__ void async_cp16(const unsigned char* g,
                                           unsigned char* l) {
  // GLOBAL_LOAD_ASYNC_TO_LDS_B128: one 16B element per lane, ASYNCcnt-tracked
  __builtin_amdgcn_global_load_async_to_lds_b128((g_i32x4*)g, (l_i32x4*)l,
                                                 0, 0);
}
#endif

// ------------------------------- helpers -----------------------------------

__device__ __forceinline__ unsigned char f32_to_fp8(float f) {
  union { float f; unsigned u; } v; v.f = f;
  unsigned u = v.u;
  unsigned s = (u >> 24) & 0x80u;
  int e = (int)((u >> 23) & 0xFFu) - 127;
  unsigned m = (u >> 20) & 0x7u;
  if (u & 0x00080000u) { m += 1u; if (m == 8u) { m = 0u; e += 1; } } // RNE-ish
  int eb = e + 7;                      // E4M3 bias
  if (eb <= 0)  return (unsigned char)s;            // flush to zero
  if (eb >= 16) return (unsigned char)(s | 0x7Eu);  // clamp to +-448
  return (unsigned char)(s | ((unsigned)eb << 3) | m);
}

__device__ __forceinline__ float sigm(float x) {
  return 1.0f / (1.0f + __expf(-x));
}

// fp8 WMMA fragment loader (A 16x64 or B 64x16 from row-major [row][k] bytes).
// p = base + row*K + kchunk + 8*(lane>>4).  ISA 8-bit layout: dword v holds
// K offsets {0,4,16,20,32,36,48,52} relative to the lane's K base.
__device__ __forceinline__ v8i load_frag(const unsigned char* p) {
  v8i a;
  a[0] = *(const int*)(p +  0); a[1] = *(const int*)(p +  4);
  a[2] = *(const int*)(p + 16); a[3] = *(const int*)(p + 20);
  a[4] = *(const int*)(p + 32); a[5] = *(const int*)(p + 36);
  a[6] = *(const int*)(p + 48); a[7] = *(const int*)(p + 52);
  return a;
}

// ------------------------- small conversion kernels -------------------------

__global__ void k_f32_to_fp8(const float* __restrict__ s,
                             unsigned char* __restrict__ d, int n) {
  int i = blockIdx.x * 256 + threadIdx.x;
  if (i < n) d[i] = f32_to_fp8(s[i]);
}

__global__ void k_bias_add(const float* __restrict__ a,
                           const float* __restrict__ b,
                           float* __restrict__ o, int n) {
  int i = blockIdx.x * 256 + threadIdx.x;
  if (i < n) o[i] = a[i] + b[i];
}

// pad act head (28x128 -> 64x128 fp8, bias 28 -> 64)
__global__ void k_pad_act(const float* __restrict__ W, const float* __restrict__ b,
                          unsigned char* __restrict__ Wp, float* __restrict__ bp) {
  int i = blockIdx.x * 256 + threadIdx.x;
  if (i >= 64 * HID) return;
  int r = i >> 7, c = i & 127;
  Wp[i] = (r < 28) ? f32_to_fp8(W[r * HID + c]) : (unsigned char)0;
  if (c == 0) bp[r] = (r < 28) ? b[r] : 0.0f;
}

// ------------------------- input projection kernels -------------------------
// x = relu(w2l_W @ [word_emb; pos_emb] + b), written (fp8) into the 4 streams
// that consume token inputs, time-major (row = t, then b, then d).

__global__ void k_proj_tokens(const float* __restrict__ wW,  // 128 x 44
                              const float* __restrict__ wb,
                              const float* __restrict__ word_embs,
                              const float* __restrict__ pos_embs,
                              const int* __restrict__ word_ids,
                              const int* __restrict__ pos_ids,
                              unsigned char* __restrict__ xq_buffer,
                              unsigned char* __restrict__ xq_stack,
                              unsigned char* __restrict__ xq_cfwd,
                              unsigned char* __restrict__ xq_cbwd) {
  size_t idx = (size_t)blockIdx.x * 256 + threadIdx.x;
  if (idx >= (size_t)LL * BB * HID) return;
  int d = (int)(idx & 127);
  int b = (int)((idx >> 7) & 63);
  int l = (int)(idx >> 13);
  const float* we = word_embs + (size_t)word_ids[b * LL + l] * 32;
  const float* pe = pos_embs  + (size_t)pos_ids [b * LL + l] * 12;
  const float* w  = wW + (size_t)d * 44;
  float s = wb[d];
  #pragma unroll 8
  for (int k = 0; k < 32; ++k) s += w[k] * we[k];
  #pragma unroll
  for (int k = 0; k < 12; ++k) s += w[32 + k] * pe[k];
  unsigned char q = f32_to_fp8(fmaxf(s, 0.0f));
  size_t o = (size_t)b * HID + d;
  xq_stack [(size_t)(l + 2)  * (BB * HID) + o] = q;  // [guard, nt, x...]
  xq_cfwd  [(size_t)(l + 1)  * (BB * HID) + o] = q;  // [nt, x...]
  xq_buffer[(size_t)(LL - l) * (BB * HID) + o] = q;  // [guard, x reversed]
  xq_cbwd  [(size_t)(LL - l) * (BB * HID) + o] = q;  // [nt, x reversed]
}

__global__ void k_proj_actions(const float* __restrict__ aW,  // 128 x 16
                               const float* __restrict__ ab,
                               const float* __restrict__ action_embs,
                               const int* __restrict__ action_ids,
                               unsigned char* __restrict__ xq_hist) {
  size_t idx = (size_t)blockIdx.x * 256 + threadIdx.x;
  if (idx >= (size_t)TT * BB * HID) return;
  int d = (int)(idx & 127);
  int b = (int)((idx >> 7) & 63);
  int t = (int)(idx >> 13);
  const float* ae = action_embs + (size_t)action_ids[b * TT + t] * 16;
  const float* w  = aW + (size_t)d * 16;
  float s = ab[d];
  #pragma unroll
  for (int k = 0; k < 16; ++k) s += w[k] * ae[k];
  xq_hist[(size_t)(t + 1) * (BB * HID) + (size_t)b * HID + d] =
      f32_to_fp8(fmaxf(s, 0.0f));
}

// guard rows + nt_in rows
__global__ void k_fill_heads(const float* __restrict__ ntW,  // 128 x 60
                             const float* __restrict__ ntb,
                             const float* __restrict__ nt_embs,
                             const int* __restrict__ nt_ids,
                             const float* __restrict__ g_buf,
                             const float* __restrict__ g_hist,
                             const float* __restrict__ g_stack,
                             unsigned char* __restrict__ xq_buffer,
                             unsigned char* __restrict__ xq_hist,
                             unsigned char* __restrict__ xq_stack,
                             unsigned char* __restrict__ xq_cfwd,
                             unsigned char* __restrict__ xq_cbwd) {
  int i = blockIdx.x * 256 + threadIdx.x;
  if (i >= BB * HID) return;
  int b = i >> 7, d = i & 127;
  const float* ne = nt_embs + (size_t)nt_ids[b] * 60;
  const float* w  = ntW + (size_t)d * 60;
  float s = ntb[d];
  #pragma unroll 6
  for (int k = 0; k < 60; ++k) s += w[k] * ne[k];
  unsigned char ntq = f32_to_fp8(fmaxf(s, 0.0f));
  size_t o = (size_t)b * HID + d;
  xq_stack[(size_t)(BB * HID) + o] = ntq;  // row 1
  xq_cfwd[o]  = ntq;                       // row 0
  xq_cbwd[o]  = ntq;                       // row 0
  xq_buffer[o] = f32_to_fp8(g_buf[d]);
  xq_hist[o]   = f32_to_fp8(g_hist[d]);
  xq_stack[o]  = f32_to_fp8(g_stack[d]);
}

// ------------------------- persistent LSTM kernel ---------------------------
// One workgroup (512 thr = 16 waves) per stream. LDS: Wih0|Whh0|Wih1|Whh1
// (fp8, 4x64KB) + h0,h1 (fp8, 2x8KB) + x double-buffer (2x8KB).
// Per step/layer: gates(64x512) = x@Wih^T + h@Whh^T + bias, via fp8 WMMA.
// Wave w owns quadruples q in {2w,2w+1}; quadruple q = tiles (mt, j+8*gi),
// mt=q>>3, j=q&7, so each thread holds matching i/f/g/o accumulator elements.

struct StreamArgs {
  const unsigned char* xseq;   // S x 64 x 128 fp8, time-major
  const unsigned char* w[4];   // wih0, whh0, wih1, whh1: fp8 512x128 row-major
  const float* bias0;          // 512 (bih+bhh)
  const float* bias1;
  const float* h0;             // 2x128 or null
  const float* c0;             // 2x128 or null
  float* hs_out;               // S x 64 x 128 f32 (layer-1 h per step)
  float* t0_save;              // [h0|h1|c0|c1] x (64x128) f32, or null
  int S;
  int pad;
};
struct LstmLaunch { StreamArgs st[5]; };

#define W_LDS_BYTES (4 * GATES * HID)          // 262144
#define H_LDS_BYTES (2 * BB * HID)             // 16384
#define XBUF_OFF    (W_LDS_BYTES + H_LDS_BYTES)
#if USE_ASYNC_X
#define LSTM_LDS    (W_LDS_BYTES + H_LDS_BYTES + 2 * BB * HID)  // 294912
#else
#define LSTM_LDS    (W_LDS_BYTES + H_LDS_BYTES)                 // 278528
#endif

__global__ __launch_bounds__(512) void k_lstm(LstmLaunch La) {
  const StreamArgs a = La.st[blockIdx.x];
  extern __shared__ unsigned char smem[];      // [4 weight mats][h0][h1][xbuf]
  const int tid = threadIdx.x;

  // stage all recurrent weights into LDS (16384 uint4)
  {
    uint4* dst = (uint4*)smem;
    for (int i = tid; i < 16384; i += 512)
      dst[i] = ((const uint4*)a.w[i >> 12])[i & 4095];
  }
#if USE_ASYNC_X
  // kick off async staging of x_0 into LDS buffer 0 (one b128 per lane)
  async_cp16(a.xseq + tid * 16, smem + XBUF_OFF + tid * 16);
#endif
  // init h state (broadcast initial h over batch rows)
  for (int i = tid; i < BB * HID; i += 512) {
    int n = i & 127;
    smem[W_LDS_BYTES + i]            = f32_to_fp8(a.h0 ? a.h0[n]       : 0.0f);
    smem[W_LDS_BYTES + BB*HID + i]   = f32_to_fp8(a.h0 ? a.h0[HID + n] : 0.0f);
  }

  const int wave   = tid >> 5;
  const int lane   = tid & 31;
  const int lane15 = lane & 15;
  const int hi     = lane >> 4;
  const int qbase  = wave * 2;
  const int mt     = qbase >> 3;     // shared by both quads of this wave
  const int j0     = qbase & 7;
  const int rowm   = mt * 16 + lane15;

  // register-resident cell state: creg[layer][q][r]
  v8f creg[2][2];
  #pragma unroll
  for (int l = 0; l < 2; ++l)
    #pragma unroll
    for (int q = 0; q < 2; ++q) {
      float cv = a.c0 ? a.c0[l * HID + (j0 + q) * 16 + lane15] : 0.0f;
      #pragma unroll
      for (int r = 0; r < 8; ++r) creg[l][q][r] = cv;
    }
#if USE_ASYNC_X
  __builtin_amdgcn_s_wait_asynccnt(0);   // x_0 landed
#endif
  __syncthreads();

  for (int t = 0; t < a.S; ++t) {
#if USE_ASYNC_X
    // overlap: start pulling x_{t+1} into the other LDS buffer now
    if (t + 1 < a.S)
      async_cp16(a.xseq + (size_t)(t + 1) * (BB * HID) + tid * 16,
                 smem + XBUF_OFF + ((t + 1) & 1) * (BB * HID) + tid * 16);
    const unsigned char* xt = smem + XBUF_OFF + (t & 1) * (BB * HID);
#else
    const unsigned char* xt = a.xseq + (size_t)t * (BB * HID);
    if (t + 1 < a.S)
      __builtin_prefetch(a.xseq + (size_t)(t + 1) * (BB * HID), 0, 1);
#endif

    #pragma unroll
    for (int layer = 0; layer < 2; ++layer) {
      const float* bias = layer ? a.bias1 : a.bias0;
      const unsigned char* wih = smem + (layer ? 2 : 0) * (GATES * HID);
      const unsigned char* whh = smem + (layer ? 3 : 1) * (GATES * HID);
      const unsigned char* hin = smem + W_LDS_BYTES + layer * (BB * HID);

      // gate accumulators, init with bias splat
      v8f g[2][4];
      #pragma unroll
      for (int q = 0; q < 2; ++q)
        #pragma unroll
        for (int gi = 0; gi < 4; ++gi) {
          float bv = bias[gi * HID + (j0 + q) * 16 + lane15];
          #pragma unroll
          for (int r = 0; r < 8; ++r) g[q][gi][r] = bv;
        }

      // GEMM: x@Wih^T + h@Whh^T  (K = 128 = 2 chunks of 64)
      #pragma unroll
      for (int kc = 0; kc < HID; kc += 64) {
        v8i ax;
        if (layer == 0)
          ax = load_frag(xt + rowm * HID + kc + hi * 8);
        else
          ax = load_frag(smem + W_LDS_BYTES + rowm * HID + kc + hi * 8); // h0
        v8i ah = load_frag(hin + rowm * HID + kc + hi * 8);
        #pragma unroll
        for (int q = 0; q < 2; ++q) {
          #pragma unroll
          for (int gi = 0; gi < 4; ++gi) {
            int wrow = gi * HID + (j0 + q) * 16 + lane15;
            v8i bx = load_frag(wih + wrow * HID + kc + hi * 8);
            g[q][gi] = __builtin_amdgcn_wmma_f32_16x16x64_fp8_fp8(
                ax, bx, (short)0, g[q][gi], false, false);
            v8i bh = load_frag(whh + wrow * HID + kc + hi * 8);
            g[q][gi] = __builtin_amdgcn_wmma_f32_16x16x64_fp8_fp8(
                ah, bh, (short)0, g[q][gi], false, false);
          }
        }
      }
      __syncthreads();   // everyone done reading h state for this layer

      // nonlinearities; update c in registers; write new h (fp8) to LDS
      unsigned char* hout = smem + W_LDS_BYTES + layer * (BB * HID);
      #pragma unroll
      for (int q = 0; q < 2; ++q) {
        int n = (j0 + q) * 16 + lane15;
        #pragma unroll
        for (int r = 0; r < 8; ++r) {
          int m = mt * 16 + r + hi * 8;
          float iv = sigm(g[q][0][r]);
          float fv = sigm(g[q][1][r]);
          float gv = tanhf(g[q][2][r]);
          float ov = sigm(g[q][3][r]);
          float c  = fv * creg[layer][q][r] + iv * gv;
          creg[layer][q][r] = c;
          float h = ov * tanhf(c);
          hout[m * HID + n] = f32_to_fp8(h);
          if (layer == 1)
            a.hs_out[((size_t)t * BB + m) * HID + n] = h;
          if (t == 0 && a.t0_save) {
            a.t0_save[layer * (BB * HID) + m * HID + n]                 = h;
            a.t0_save[2 * (BB * HID) + layer * (BB * HID) + m * HID + n] = c;
          }
        }
      }
      __syncthreads();   // new h visible before anyone reads it
    }

#if USE_ASYNC_X
    // x_{t+1} staging must be complete (and WG-visible) before next step
    __builtin_amdgcn_s_wait_asynccnt(0);
    __syncthreads();
#endif
  }
}

// ------------------- composition + extra stack step (small) -----------------

__global__ __launch_bounds__(128) void k_comp_step(
    const float* __restrict__ comp_W, const float* __restrict__ comp_b,
    const float* __restrict__ fwd_hs, const float* __restrict__ bwd_hs, int Sc,
    const float* __restrict__ Wih0, const float* __restrict__ Whh0,
    const float* __restrict__ bih0, const float* __restrict__ bhh0,
    const float* __restrict__ Wih1, const float* __restrict__ Whh1,
    const float* __restrict__ bih1, const float* __restrict__ bhh1,
    const float* __restrict__ t0, float* __restrict__ stack_hs, int row_out) {
  __shared__ float comp[HID], h0n[HID];
  int b = blockIdx.x, d = threadIdx.x;
  const float* fv = fwd_hs + ((size_t)(Sc - 1) * BB + b) * HID;
  const float* bv = bwd_hs + ((size_t)(Sc - 1) * BB + b) * HID;
  const float* cw = comp_W + (size_t)d * (2 * HID);
  float s = comp_b[d];
  for (int k = 0; k < HID; ++k) s += cw[k] * fv[k];
  for (int k = 0; k < HID; ++k) s += cw[HID + k] * bv[k];
  comp[d] = fmaxf(s, 0.0f);
  __syncthreads();

  // layer 0
  const float* h0p = t0 + (size_t)b * HID;
  const float* c0p = t0 + 2 * (BB * HID) + (size_t)b * HID;
  float gg[4];
  #pragma unroll
  for (int gi = 0; gi < 4; ++gi) {
    int gr = gi * HID + d;
    float acc = bih0[gr] + bhh0[gr];
    const float* wi = Wih0 + (size_t)gr * HID;
    const float* wh = Whh0 + (size_t)gr * HID;
    for (int k = 0; k < HID; ++k) acc += wi[k] * comp[k] + wh[k] * h0p[k];
    gg[gi] = acc;
  }
  float c0v = sigm(gg[1]) * c0p[d] + sigm(gg[0]) * tanhf(gg[2]);
  h0n[d] = sigm(gg[3]) * tanhf(c0v);
  __syncthreads();

  // layer 1
  const float* h1p = t0 + (BB * HID) + (size_t)b * HID;
  const float* c1p = t0 + 3 * (BB * HID) + (size_t)b * HID;
  #pragma unroll
  for (int gi = 0; gi < 4; ++gi) {
    int gr = gi * HID + d;
    float acc = bih1[gr] + bhh1[gr];
    const float* wi = Wih1 + (size_t)gr * HID;
    const float* wh = Whh1 + (size_t)gr * HID;
    for (int k = 0; k < HID; ++k) acc += wi[k] * h0n[k] + wh[k] * h1p[k];
    gg[gi] = acc;
  }
  float c1v = sigm(gg[1]) * c1p[d] + sigm(gg[0]) * tanhf(gg[2]);
  float h1v = sigm(gg[3]) * tanhf(c1v);
  stack_hs[((size_t)row_out * BB + b) * HID + d] = h1v;
}

// --------------------------- feature gather ---------------------------------
// feats (fp8, row = b*1027 + i, 384 cols): [stack_top | buf_top[bidx] | hist]

__global__ void k_gather_feats(const float* __restrict__ stack_hs,
                               const float* __restrict__ buf_hs,
                               const float* __restrict__ hist_hs,
                               unsigned char* __restrict__ feats) {
  size_t idx = (size_t)blockIdx.x * 256 + threadIdx.x;
  if (idx >= (size_t)BB * NTOP * HID) return;
  int d = (int)(idx & 127);
  int i = (int)((idx >> 7) % NTOP);
  int b = (int)(idx / ((size_t)NTOP * HID));
  size_t row = (size_t)b * NTOP + i;
  int bi = (i == 0) ? LL : ((i <= LL) ? (LL + 1 - i) : 0);
  float sv = stack_hs[((size_t)i  * BB + b) * HID + d];
  float bv = buf_hs  [((size_t)bi * BB + b) * HID + d];
  float hv = hist_hs [((size_t)i  * BB + b) * HID + d];
  feats[row * 384 +       d] = f32_to_fp8(sv);
  feats[row * 384 + 128 + d] = f32_to_fp8(bv);
  feats[row * 384 + 256 + d] = f32_to_fp8(hv);
}

// --------------------------- generic fp8 WMMA GEMM --------------------------
// C[M,N] = relu?(A[M,K] @ W[N,K]^T + bias).  Block tile 64x64, 8 waves,
// 2 tiles per wave (shared A fragment), K in chunks of 64.

__global__ __launch_bounds__(256) void k_gemm_fp8(
    const unsigned char* __restrict__ A, int K,
    const unsigned char* __restrict__ W,
    const float* __restrict__ bias,
    float* __restrict__ C, int N, int relu) {
  int mb = blockIdx.x, nb = blockIdx.y;
  int wave = threadIdx.x >> 5, lane = threadIdx.x & 31;
  int lane15 = lane & 15, hi = lane >> 4;
  int mt  = wave & 3;
  int ntb = (wave >> 2) * 2;

  v8f acc[2];
  #pragma unroll
  for (int q = 0; q < 2; ++q) {
    float bv = bias ? bias[nb * 64 + (ntb + q) * 16 + lane15] : 0.0f;
    #pragma unroll
    for (int r = 0; r < 8; ++r) acc[q][r] = bv;
  }

  int rowA = mb * 64 + mt * 16 + lane15;
  for (int kc = 0; kc < K; kc += 64) {
    v8i af = load_frag(A + (size_t)rowA * K + kc + hi * 8);
    #pragma unroll
    for (int q = 0; q < 2; ++q) {
      int rowW = nb * 64 + (ntb + q) * 16 + lane15;
      v8i bf = load_frag(W + (size_t)rowW * K + kc + hi * 8);
      acc[q] = __builtin_amdgcn_wmma_f32_16x16x64_fp8_fp8(
          af, bf, (short)0, acc[q], false, false);
    }
  }

  #pragma unroll
  for (int q = 0; q < 2; ++q) {
    int gn = nb * 64 + (ntb + q) * 16 + lane15;
    #pragma unroll
    for (int r = 0; r < 8; ++r) {
      int gm = mb * 64 + mt * 16 + r + hi * 8;
      float v = acc[q][r];
      if (relu) v = fmaxf(v, 0.0f);
      C[(size_t)gm * N + gn] = v;
    }
  }
}

// ------------------------------ log_softmax ---------------------------------

__global__ void k_logsoftmax(const float* __restrict__ logits,  // M x 64 (28 used)
                             float* __restrict__ out, int M) {
  int row = blockIdx.x * 256 + threadIdx.x;
  if (row >= M) return;
  const float* l = logits + (size_t)row * 64;
  float mx = l[0];
  for (int j = 1; j < 28; ++j) mx = fmaxf(mx, l[j]);
  float s = 0.0f;
  for (int j = 0; j < 28; ++j) s += __expf(l[j] - mx);
  float lse = mx + __logf(s);
  for (int j = 0; j < 28; ++j) out[(size_t)row * 28 + j] = l[j] - lse;
}

// ------------------------------- host side ----------------------------------

static inline unsigned char* bump(unsigned char*& p, size_t bytes) {
  unsigned char* r = p;
  p += (bytes + 255) & ~(size_t)255;
  return r;
}

extern "C" void kernel_launch(void* const* d_in, const int* in_sizes, int n_in,
                              void* d_out, int out_size, void* d_ws, size_t ws_size,
                              hipStream_t stream) {
  (void)in_sizes; (void)n_in; (void)out_size; (void)ws_size;

  // Input flattening assumed to follow jax tree order (sorted dict keys):
  // 0 action_ids, 1 nt_ids, 2.. params leaves (sorted), then pos_ids, word_ids.
  const int* action_ids = (const int*)d_in[0];
  const int* nt_ids     = (const int*)d_in[1];
  auto P = [&](int leaf) -> const float* { return (const float*)d_in[2 + leaf]; };
  const int* pos_ids    = (const int*)d_in[67];
  const int* word_ids   = (const int*)d_in[68];

  // params leaf indices (sorted keys; *_layers: [Whh,Wih,bhh,bih] x 2 layers)
  enum {
    La2l_W = 0, La2l_b, Lact_W, Lact_b, Laction_embs,
    Lbuffer_c0, Lbuffer_guard, Lbuffer_h0,
    Lbuf_lay = 8,    // 8..15
    Lcbwd_lay = 16,  // 16..23
    Lcfwd_lay = 24,  // 24..31
    Lcomp_W = 32, Lcomp_b = 33,
    Lhist_c0 = 34, Lhist_guard = 35, Lhist_h0 = 36,
    Lhist_lay = 37,  // 37..44
    Lnt2l_W = 45, Lnt2l_b = 46, Lnt_embs = 47, Lpos_embs = 48,
    Lstack_c0 = 49, Lstack_guard = 50, Lstack_h0 = 51,
    Lstack_lay = 52, // 52..59
    Lsumm_W = 60, Lsumm_b = 61, Lw2l_W = 62, Lw2l_b = 63, Lword_embs = 64
  };

  // streams: 0=buffer 1=history 2=stack 3=cfwd 4=cbwd
  const int layBase[5] = { Lbuf_lay, Lhist_lay, Lstack_lay, Lcfwd_lay, Lcbwd_lay };
  const int h0Leaf[5]  = { Lbuffer_h0, Lhist_h0, Lstack_h0, -1, -1 };
  const int c0Leaf[5]  = { Lbuffer_c0, Lhist_c0, Lstack_c0, -1, -1 };
  const int Svals[5]   = { LL + 1, TT + 1, LL + 2, LL + 1, LL + 1 }; // 1025,1027,1026,1025,1025
  const int hsRows[5]  = { LL + 1, TT + 1, LL + 3, LL + 1, LL + 1 }; // stack gets +1 row for h_r

  unsigned char* wp = (unsigned char*)d_ws;

  unsigned char* xq[5];
  float* hs[5];
  unsigned char* wih8[5][2]; unsigned char* whh8[5][2]; float* biasc[5][2];
  for (int s = 0; s < 5; ++s) {
    xq[s] = bump(wp, (size_t)Svals[s] * BB * HID);
    hs[s] = (float*)bump(wp, (size_t)hsRows[s] * BB * HID * sizeof(float));
    for (int l = 0; l < 2; ++l) {
      whh8[s][l] = bump(wp, GATES * HID);
      wih8[s][l] = bump(wp, GATES * HID);
      biasc[s][l] = (float*)bump(wp, GATES * sizeof(float));
    }
  }
  float*         t0_save  = (float*)bump(wp, 4 * BB * HID * sizeof(float));
  unsigned char* summ8    = bump(wp, HID * 384);
  unsigned char* act8     = bump(wp, 64 * HID);
  float*         act_bp   = (float*)bump(wp, 64 * sizeof(float));
  unsigned char* feats    = bump(wp, (size_t)MROWS * 384);
  float*         summary  = (float*)bump(wp, (size_t)MROWS * HID * sizeof(float));
  unsigned char* summary8 = bump(wp, (size_t)MROWS * HID);
  float*         logits   = (float*)bump(wp, (size_t)MROWS * 64 * sizeof(float));

  // --- weight conversion: f32 -> fp8 row-major; bias = bih + bhh
  for (int s = 0; s < 5; ++s)
    for (int l = 0; l < 2; ++l) {
      int base = layBase[s] + l * 4;
      k_f32_to_fp8<<<(GATES * HID + 255) / 256, 256, 0, stream>>>(
          P(base + 0), whh8[s][l], GATES * HID);
      k_f32_to_fp8<<<(GATES * HID + 255) / 256, 256, 0, stream>>>(
          P(base + 1), wih8[s][l], GATES * HID);
      k_bias_add<<<2, 256, 0, stream>>>(P(base + 3), P(base + 2),
                                        biasc[s][l], GATES);
    }
  k_f32_to_fp8<<<(HID * 384 + 255) / 256, 256, 0, stream>>>(
      P(Lsumm_W), summ8, HID * 384);
  k_pad_act<<<(64 * HID + 255) / 256, 256, 0, stream>>>(
      P(Lact_W), P(Lact_b), act8, act_bp);

  // --- input projections into fp8 time-major stream buffers
  {
    size_t tot = (size_t)LL * BB * HID;
    k_proj_tokens<<<(unsigned)((tot + 255) / 256), 256, 0, stream>>>(
        P(Lw2l_W), P(Lw2l_b), P(Lword_embs), P(Lpos_embs),
        word_ids, pos_ids, xq[0], xq[2], xq[3], xq[4]);
  }
  {
    size_t tot = (size_t)TT * BB * HID;
    k_proj_actions<<<(unsigned)((tot + 255) / 256), 256, 0, stream>>>(
        P(La2l_W), P(La2l_b), P(Laction_embs), action_ids, xq[1]);
  }
  k_fill_heads<<<(BB * HID + 255) / 256, 256, 0, stream>>>(
      P(Lnt2l_W), P(Lnt2l_b), P(Lnt_embs), nt_ids,
      P(Lbuffer_guard), P(Lhist_guard), P(Lstack_guard),
      xq[0], xq[1], xq[2], xq[3], xq[4]);

  // --- persistent LSTM scan: 5 concurrent WGPs, fp8 WMMA, LDS weights
  LstmLaunch launch;
  for (int s = 0; s < 5; ++s) {
    StreamArgs& A = launch.st[s];
    A.xseq = xq[s];
    A.w[0] = wih8[s][0]; A.w[1] = whh8[s][0];
    A.w[2] = wih8[s][1]; A.w[3] = whh8[s][1];
    A.bias0 = biasc[s][0]; A.bias1 = biasc[s][1];
    A.h0 = (h0Leaf[s] >= 0) ? P(h0Leaf[s]) : nullptr;
    A.c0 = (c0Leaf[s] >= 0) ? P(c0Leaf[s]) : nullptr;
    A.hs_out = hs[s];
    A.t0_save = (s == 2) ? t0_save : nullptr;
    A.S = Svals[s];
    A.pad = 0;
  }
  k_lstm<<<5, 512, LSTM_LDS, stream>>>(launch);

  // --- composed = relu(comp_W @ [fwd_final; bwd_final] + b); extra stack step
  {
    int sb = Lstack_lay;
    k_comp_step<<<BB, HID, 0, stream>>>(
        P(Lcomp_W), P(Lcomp_b), hs[3], hs[4], Svals[3],
        P(sb + 1), P(sb + 0), P(sb + 3), P(sb + 2),
        P(sb + 5), P(sb + 4), P(sb + 7), P(sb + 6),
        t0_save, hs[2], LL + 2 /* row 1026 */);
  }

  // --- feats gather -> summ GEMM -> act GEMM -> log_softmax
  {
    size_t tot = (size_t)BB * NTOP * HID;
    k_gather_feats<<<(unsigned)((tot + 255) / 256), 256, 0, stream>>>(
        hs[2], hs[0], hs[1], feats);
  }
  {
    dim3 g(MROWS / 64, HID / 64);
    k_gemm_fp8<<<g, 256, 0, stream>>>(feats, 384, summ8, P(Lsumm_b),
                                      summary, HID, 1);
  }
  k_f32_to_fp8<<<(unsigned)(((size_t)MROWS * HID + 255) / 256), 256, 0, stream>>>(
      summary, summary8, MROWS * HID);
  {
    dim3 g(MROWS / 64, 1);
    k_gemm_fp8<<<g, 256, 0, stream>>>(summary8, HID, act8, act_bp,
                                      logits, 64, 0);
  }
  k_logsoftmax<<<(MROWS + 255) / 256, 256, 0, stream>>>(logits, (float*)d_out,
                                                        MROWS);
}
